// GUPPredictor_14113262535327
// MI455X (gfx1250) — matrix operand order repaired and checked by reading, not compiled, not run on hardware
//
#include <hip/hip_runtime.h>
#include <math.h>

typedef _Float16 v16h __attribute__((ext_vector_type(16)));
typedef _Float16 v8h  __attribute__((ext_vector_type(8)));
typedef float    v8f  __attribute__((ext_vector_type(8)));

#define USE_ASYNC_LDS 1

#define B_    8
#define CIN   64
#define H_    96
#define W_    320
#define HW    (H_*W_)
#define HC    256
#define KDET  50
#define NROI  400

// ---- main conv k-space: k = r*64 + c, K = 576, 18 k-steps of 32
#define MAIN_KSTEPS 18
#define PXB   64            // output pixels per block (along W)
#define PCOLS (PXB + 2)     // 66 patch columns
#define PCH   72            // padded channel stride (16B aligned, conflict-free banks)

// ---- roi conv k-space: k = r*72 + c (channels padded 69->72), K = 648, 21 k-steps
#define ROI_CPAD   72
#define ROI_KSTEPS 21
#define ROI_SLAB   (81*72)   // 9x9 spatial (padded) x 72 ch, f16

// output offsets (floats)
#define OUT_HM 0
#define OUT_O2 737280
#define OUT_S2 1228800
#define OUT_HD 1720320
#define OUT_DEP 1729920
#define OUT_O3 1730720
#define OUT_S3 1731520
#define OUT_H3 1732720

#if USE_ASYNC_LDS
__device__ __forceinline__ void async_ld_b128(void* lds_ptr, const void* gptr) {
  unsigned lo = (unsigned)(uintptr_t)lds_ptr;  // LDS aperture: low 32 bits = LDS offset
  asm volatile("global_load_async_to_lds_b128 %0, %1, off"
               :: "v"(lo), "v"((unsigned long long)(uintptr_t)gptr)
               : "memory");
}
__device__ __forceinline__ void async_wait0() {
  asm volatile("s_wait_asynccnt 0" ::: "memory");
}
#endif

// ---------------------------------------------------------------------------
// features NCHW f32 -> NHWC f16
__global__ void k_feat_to_f16(const float* __restrict__ src, _Float16* __restrict__ dst) {
  int idx = blockIdx.x * 256 + threadIdx.x;
  if (idx >= B_ * CIN * H_ * W_) return;
  int x = idx % W_;
  int t = idx / W_;
  int y = t % H_; t /= H_;
  int c = t % CIN;
  int b = t / CIN;
  dst[(((b * H_) + y) * W_ + x) * CIN + c] = (_Float16)src[idx];
}

// ---------------------------------------------------------------------------
// pack conv3x3 weights (O=256, Cin, 3, 3) into per-lane WMMA B-fragment layout:
// idx = ((nt*ksteps + kk)*32 + lane)*16 + e ; k = kk*32 + (lane<16?0:8) + (e<8 ? e : 8+e) ;
// n = nt*16 + lane%16 ; k = r*CinPad + c
__global__ void k_pack_w(const float* __restrict__ w1, int Cin, int CinPad, int ksteps,
                         _Float16* __restrict__ out) {
  int idx = blockIdx.x * 256 + threadIdx.x;
  int total = 16 * ksteps * 512;
  if (idx >= total) return;
  int e    = idx & 15;
  int lane = (idx >> 4) & 31;
  int kk   = (idx >> 9) % ksteps;
  int nt   = idx / (512 * ksteps);
  int k = kk * 32 + ((lane < 16) ? 0 : 8) + ((e < 8) ? e : 8 + e);
  int n = nt * 16 + (lane & 15);
  float v = 0.f;
  int Ktot = 9 * CinPad;
  if (k < Ktot) {
    int r = k / CinPad;
    int c = k % CinPad;
    if (c < Cin) v = w1[(n * Cin + c) * 9 + r];  // w1[n][c][dy][dx], r=dy*3+dx
  }
  out[idx] = (_Float16)v;
}

// ---------------------------------------------------------------------------
// fused branch2: conv3x3(64->256)+bias+ReLU then conv1x1(256->oc)+bias
// one block = 64 output pixels (one row) x 256 channels; 8 waves,
// each wave: 4 m-tiles x 2 n-tiles WMMA. A-fragments read directly from the
// raw NHWC patch in LDS (no im2col materialization).
__global__ __launch_bounds__(256) void k_branch(const _Float16* __restrict__ feat16,
                                                const _Float16* __restrict__ wpack,
                                                const float* __restrict__ b1,
                                                const float* __restrict__ w2,
                                                const float* __restrict__ b2,
                                                float* __restrict__ out, int oc) {
  __shared__ __align__(16) _Float16 patch[3 * PCOLS * PCH];
  __shared__ __align__(16) _Float16 hidden[PXB * HC];
  int tid = threadIdx.x;
  int wt = blockIdx.x % (W_ / PXB);
  int h  = (blockIdx.x / (W_ / PXB)) % H_;
  int b  = blockIdx.x / ((W_ / PXB) * H_);
  int w0 = wt * PXB;

  // stage raw patch: rows h-1..h+1, cols w0-1..w0+PXB, 64 ch (f16, NHWC-contig)
  v8h zero8 = {};
  for (int i = tid; i < 3 * PCOLS * 8; i += 256) {
    int j  = i & 7;
    int x  = (i >> 3) % PCOLS;
    int dy = i / (PCOLS * 8);
    int gy = h + dy - 1;
    int gx = w0 + x - 1;
    _Float16* dst = &patch[(dy * PCOLS + x) * PCH + j * 8];
    if (gy >= 0 && gy < H_ && gx >= 0 && gx < W_) {
      const _Float16* src = &feat16[(((b * H_) + gy) * W_ + gx) * CIN + j * 8];
#if USE_ASYNC_LDS
      async_ld_b128(dst, src);
#else
      *(v8h*)dst = *(const v8h*)src;
#endif
    } else {
      *(v8h*)dst = zero8;
    }
  }
#if USE_ASYNC_LDS
  async_wait0();
#endif
  __syncthreads();

  int lane = tid & 31;
  int wv = tid >> 5;
  int mloc = lane & 15;
  int aoff1 = (lane < 16) ? 0 : 8;
  int aoff2 = (lane < 16) ? 16 : 24;
  int nt0 = wv * 2, nt1 = nt0 + 1;
  const _Float16* wp0 = wpack + nt0 * MAIN_KSTEPS * 512;
  const _Float16* wp1 = wpack + nt1 * MAIN_KSTEPS * 512;

  v8f acc[4][2];
  v8f zf = {};
#pragma unroll
  for (int mt = 0; mt < 4; ++mt) { acc[mt][0] = zf; acc[mt][1] = zf; }

  for (int kk = 0; kk < MAIN_KSTEPS; ++kk) {
    v16h bf0 = *(const v16h*)(wp0 + (kk * 32 + lane) * 16);
    v16h bf1 = *(const v16h*)(wp1 + (kk * 32 + lane) * 16);
    int k1 = kk * 32 + aoff1;
    int k2 = kk * 32 + aoff2;
    int r1 = k1 >> 6, c1 = k1 & 63;
    int r2 = k2 >> 6, c2 = k2 & 63;
    int o1 = ((r1 / 3) * PCOLS + (r1 % 3)) * PCH + c1;
    int o2 = ((r2 / 3) * PCOLS + (r2 % 3)) * PCH + c2;
#pragma unroll
    for (int mt = 0; mt < 4; ++mt) {
      int m = mt * 16 + mloc;
      v8h lo = *(const v8h*)&patch[o1 + m * PCH];
      v8h hi = *(const v8h*)&patch[o2 + m * PCH];
      v16h a;
#pragma unroll
      for (int q = 0; q < 8; ++q) { a[q] = lo[q]; a[8 + q] = hi[q]; }
      acc[mt][0] = __builtin_amdgcn_wmma_f32_16x16x32_f16(false, a, false, bf0, (short)0, acc[mt][0], false, false);
      acc[mt][1] = __builtin_amdgcn_wmma_f32_16x16x32_f16(false, a, false, bf1, (short)0, acc[mt][1], false, false);
    }
  }

  // bias + relu -> hidden tile (f16) in LDS
  int m0 = (lane < 16) ? 0 : 8;
  int ch0 = nt0 * 16 + mloc;
  int ch1 = nt1 * 16 + mloc;
  float bb0 = b1[ch0], bb1 = b1[ch1];
#pragma unroll
  for (int mt = 0; mt < 4; ++mt) {
#pragma unroll
    for (int r = 0; r < 8; ++r) {
      float v0 = acc[mt][0][r] + bb0; v0 = v0 > 0.f ? v0 : 0.f;
      float v1 = acc[mt][1][r] + bb1; v1 = v1 > 0.f ? v1 : 0.f;
      hidden[(mt * 16 + m0 + r) * HC + ch0] = (_Float16)v0;
      hidden[(mt * 16 + m0 + r) * HC + ch1] = (_Float16)v1;
    }
  }
  __syncthreads();

  // fused 1x1 conv
  for (int mo = tid; mo < PXB * oc; mo += 256) {
    int m = mo % PXB;
    int o = mo / PXB;
    float s = b2[o];
    for (int c = 0; c < HC; ++c) s += (float)hidden[m * HC + c] * w2[o * HC + c];
    out[((b * oc + o) * H_ + h) * W_ + (w0 + m)] = s;
  }
}

// ---------------------------------------------------------------------------
__global__ void k_nms(const float* __restrict__ heat, float* __restrict__ outp) {
  int idx = blockIdx.x * 256 + threadIdx.x;
  if (idx >= B_ * 3 * HW) return;
  int x = idx % W_;
  int y = (idx / W_) % H_;
  int bc = idx / HW;
  float h = heat[idx];
  float mx = -INFINITY;
  for (int dy = -1; dy <= 1; ++dy) {
    int yy = y + dy; if (yy < 0 || yy >= H_) continue;
    for (int dx = -1; dx <= 1; ++dx) {
      int xx = x + dx; if (xx < 0 || xx >= W_) continue;
      float v = heat[bc * HW + yy * W_ + xx];
      mx = v > mx ? v : mx;
    }
  }
  outp[idx] = (mx == h) ? h : 0.f;
}

// iterative stable top-K per (b,c): replicates lax.top_k ordering
__global__ void k_topk1(const float* __restrict__ nh, float* __restrict__ s_all,
                        int* __restrict__ i_all) {
  int bc = blockIdx.x;
  const float* v = nh + bc * HW;
  __shared__ float sv[256]; __shared__ int si[256];
  __shared__ float pv; __shared__ int pi;
  int tid = threadIdx.x;
  if (tid == 0) { pv = INFINITY; pi = -1; }
  __syncthreads();
  for (int k = 0; k < KDET; ++k) {
    float bv = -INFINITY; int bi = 0x7fffffff;
    for (int i = tid; i < HW; i += 256) {
      float val = v[i];
      bool after = (val < pv) || (val == pv && i > pi);
      if (after && (val > bv || (val == bv && i < bi))) { bv = val; bi = i; }
    }
    sv[tid] = bv; si[tid] = bi;
    __syncthreads();
    for (int s = 128; s > 0; s >>= 1) {
      if (tid < s) {
        if (sv[tid + s] > sv[tid] || (sv[tid + s] == sv[tid] && si[tid + s] < si[tid])) {
          sv[tid] = sv[tid + s]; si[tid] = si[tid + s];
        }
      }
      __syncthreads();
    }
    if (tid == 0) { s_all[bc * KDET + k] = sv[0]; i_all[bc * KDET + k] = si[0]; pv = sv[0]; pi = si[0]; }
    __syncthreads();
  }
}

__global__ void k_topk2(const float* __restrict__ s_all, const int* __restrict__ i_all,
                        int* __restrict__ inds, int* __restrict__ classes) {
  int b = blockIdx.x;
  const float* v = s_all + b * 3 * KDET;
  const int* ia = i_all + b * 3 * KDET;
  __shared__ float sv[256]; __shared__ int si[256];
  __shared__ float pv; __shared__ int pi;
  int tid = threadIdx.x;
  if (tid == 0) { pv = INFINITY; pi = -1; }
  __syncthreads();
  for (int k = 0; k < KDET; ++k) {
    float bv = -INFINITY; int bi = 0x7fffffff;
    if (tid < 3 * KDET) {
      float val = v[tid];
      if ((val < pv) || (val == pv && tid > pi)) { bv = val; bi = tid; }
    }
    sv[tid] = bv; si[tid] = bi;
    __syncthreads();
    for (int s = 128; s > 0; s >>= 1) {
      if (tid < s) {
        if (sv[tid + s] > sv[tid] || (sv[tid + s] == sv[tid] && si[tid + s] < si[tid])) {
          sv[tid] = sv[tid + s]; si[tid] = si[tid + s];
        }
      }
      __syncthreads();
    }
    if (tid == 0) {
      int j = si[0];
      inds[b * KDET + k] = ia[j];
      classes[b * KDET + k] = j / KDET;
      pv = sv[0]; pi = j;
    }
    __syncthreads();
  }
}

// ---------------------------------------------------------------------------
// per-ROI scalars: feature-space box, image-space projections, coord rows/cols
// rois[n*20]: x1,y1,bw,bh, cx[7], cy[7], f_u, box2d_h
__global__ void k_rois(const int* __restrict__ inds, const float* __restrict__ off2d,
                       const float* __restrict__ s2d, const float* __restrict__ calib,
                       const float* __restrict__ coord_ranges, float* __restrict__ rois) {
  int n = blockIdx.x * 256 + threadIdx.x;
  if (n >= NROI) return;
  int b = n / KDET;
  int i = inds[n];
  int y = i / W_, x = i % W_;
  float ox = off2d[(b * 2 + 0) * HW + i];
  float oy = off2d[(b * 2 + 1) * HW + i];
  float sw = s2d[(b * 2 + 0) * HW + i];
  float sh = s2d[(b * 2 + 1) * HW + i];
  float cx = (float)x + ox, cy = (float)y + oy;
  float bx1 = cx - sw * 0.5f, by1 = cy - sh * 0.5f;
  float bx2 = cx + sw * 0.5f, by2 = cy + sh * 0.5f;
  const float* cr = coord_ranges + b * 4;
  float sx = cr[2] - cr[0], sy = cr[3] - cr[1];
  float ix1 = bx1 / (float)W_ * sx + cr[0];
  float iy1 = by1 / (float)H_ * sy + cr[1];
  float ix2 = bx2 / (float)W_ * sx + cr[0];
  float iy2 = by2 / (float)H_ * sy + cr[1];
  const float* cb = calib + b * 12;
  float f_u = cb[0], c_u = cb[2], f_v = cb[5], c_v = cb[6];
  float bxo = cb[3] / (-f_u), byo = cb[7] / (-f_v);
  float p1x = (ix1 - c_u) / f_u + bxo, p1y = (iy1 - c_v) / f_v + byo;
  float p2x = (ix2 - c_u) / f_u + bxo, p2y = (iy2 - c_v) / f_v + byo;
  float* R = rois + n * 20;
  R[0] = bx1; R[1] = by1;
  R[2] = fmaxf(bx2 - bx1, 1.f) / 7.f;
  R[3] = fmaxf(by2 - by1, 1.f) / 7.f;
  for (int j = 0; j < 7; ++j) {
    float t = (float)j / 6.f;
    R[4 + j]  = p1x + t * (p2x - p1x);
    R[11 + j] = p1y + t * (p2y - p1y);
  }
  R[18] = f_u;
  R[19] = fmaxf(iy2 - iy1, 1.f);
}

// ---------------------------------------------------------------------------
// ROI-align 7x7 (sr=2) + build padded 9x9x72 f16 head-input slab per ROI
__global__ __launch_bounds__(256) void k_roifill(const float* __restrict__ feat,
                                                 const float* __restrict__ rois,
                                                 const int* __restrict__ classes,
                                                 _Float16* __restrict__ roi_in) {
  int n = blockIdx.x;
  __shared__ float R[20];
  __shared__ int clsS;
  if (threadIdx.x < 20) R[threadIdx.x] = rois[n * 20 + threadIdx.x];
  if (threadIdx.x == 0) clsS = classes[n];
  __syncthreads();
  int b = n / KDET;
  float x1 = R[0], y1 = R[1], bw = R[2], bh = R[3];
  int cls = clsS;
  const float* fb = feat + (size_t)b * CIN * HW;
  _Float16* dst = roi_in + (size_t)n * ROI_SLAB;
  for (int i = threadIdx.x; i < ROI_SLAB; i += 256) {
    int c = i % ROI_CPAD;
    int p = i / ROI_CPAD;
    int yy = p / 9, xx = p % 9;
    float val = 0.f;
    if (yy >= 1 && yy <= 7 && xx >= 1 && xx <= 7 && c < 69) {
      int py = yy - 1, px = xx - 1;
      if (c < 64) {
        float s = 0.f;
        const float* fc = fb + (size_t)c * HW;
#pragma unroll
        for (int u = 0; u < 2; ++u) {
          float Y = y1 + bh * ((float)(py * 2 + u) + 0.5f) * 0.5f;
#pragma unroll
          for (int t2 = 0; t2 < 2; ++t2) {
            float X = x1 + bw * ((float)(px * 2 + t2) + 0.5f) * 0.5f;
            bool valid = (X > -1.f) && (X < (float)W_) && (Y > -1.f) && (Y < (float)H_);
            float Xc = fminf(fmaxf(X, 0.f), (float)(W_ - 1));
            float Yc = fminf(fmaxf(Y, 0.f), (float)(H_ - 1));
            int x0 = (int)floorf(Xc), y0 = (int)floorf(Yc);
            int x1i = min(x0 + 1, W_ - 1), y1i = min(y0 + 1, H_ - 1);
            float lx = Xc - (float)x0, ly = Yc - (float)y0;
            float v = (1.f - ly) * (1.f - lx) * fc[y0 * W_ + x0]
                    + (1.f - ly) * lx        * fc[y0 * W_ + x1i]
                    + ly        * (1.f - lx) * fc[y1i * W_ + x0]
                    + ly        * lx         * fc[y1i * W_ + x1i];
            s += valid ? v : 0.f;
          }
        }
        val = s * 0.25f;
      } else if (c == 64) val = R[4 + px];
      else if (c == 65)   val = R[11 + py];
      else                val = (cls == (c - 66)) ? 1.f : 0.f;
    }
    dst[i] = (_Float16)val;
  }
}

// ---------------------------------------------------------------------------
// fused ROI head: conv3x3(69->256)+bias, BN, ReLU, global mean, conv1x1(256->oc)
// one block per ROI; 8 waves x (4 m-tiles x 2 n-tiles) WMMA
__global__ __launch_bounds__(256) void k_roihead(const _Float16* __restrict__ roi_in,
                                                 const _Float16* __restrict__ wpack,
                                                 const float* __restrict__ b1,
                                                 const float* __restrict__ g,
                                                 const float* __restrict__ beta,
                                                 const float* __restrict__ mean,
                                                 const float* __restrict__ var,
                                                 const float* __restrict__ w2,
                                                 const float* __restrict__ b2,
                                                 float* __restrict__ out, int oc) {
  __shared__ __align__(16) _Float16 slab[ROI_SLAB];
  __shared__ float pooled[HC];
  int n = blockIdx.x;
  int tid = threadIdx.x;
  {
    const _Float16* src = roi_in + (size_t)n * ROI_SLAB;
    for (int i = tid; i < ROI_SLAB / 8; i += 256) {
#if USE_ASYNC_LDS
      async_ld_b128(&slab[i * 8], src + i * 8);
#else
      *(v8h*)&slab[i * 8] = *(const v8h*)(src + i * 8);
#endif
    }
#if USE_ASYNC_LDS
    async_wait0();
#endif
  }
  __syncthreads();

  int lane = tid & 31, wv = tid >> 5;
  int mloc = lane & 15;
  int aoff1 = (lane < 16) ? 0 : 8;
  int aoff2 = (lane < 16) ? 16 : 24;
  int nt0 = wv * 2, nt1 = nt0 + 1;
  const _Float16* wp0 = wpack + nt0 * ROI_KSTEPS * 512;
  const _Float16* wp1 = wpack + nt1 * ROI_KSTEPS * 512;

  int pyb[4], pxb[4]; bool mok[4];
#pragma unroll
  for (int mt = 0; mt < 4; ++mt) {
    int m = mt * 16 + mloc;
    mok[mt] = (m < 49);
    pyb[mt] = m / 7;
    pxb[mt] = m % 7;
  }

  v8f acc[4][2];
  v8f zero = {};
#pragma unroll
  for (int mt = 0; mt < 4; ++mt) { acc[mt][0] = zero; acc[mt][1] = zero; }

  for (int kk = 0; kk < ROI_KSTEPS; ++kk) {
    v16h bf0 = *(const v16h*)(wp0 + (kk * 32 + lane) * 16);
    v16h bf1 = *(const v16h*)(wp1 + (kk * 32 + lane) * 16);
    int kb = kk * 32;
#pragma unroll
    for (int mt = 0; mt < 4; ++mt) {
      v16h a;
      if (mok[mt]) {
        v8h lo = {}; v8h hi = {};
        {
          int k0 = kb + aoff1;
          int r = k0 / 72;
          if (r < 9) {
            int c0 = k0 - r * 72;
            lo = *(const v8h*)&slab[((pyb[mt] + r / 3) * 9 + (pxb[mt] + r % 3)) * ROI_CPAD + c0];
          }
        }
        {
          int k0 = kb + aoff2;
          int r = k0 / 72;
          if (r < 9) {
            int c0 = k0 - r * 72;
            hi = *(const v8h*)&slab[((pyb[mt] + r / 3) * 9 + (pxb[mt] + r % 3)) * ROI_CPAD + c0];
          }
        }
#pragma unroll
        for (int q = 0; q < 8; ++q) { a[q] = lo[q]; a[8 + q] = hi[q]; }
      } else {
#pragma unroll
        for (int q = 0; q < 16; ++q) a[q] = (_Float16)0.f;
      }
      acc[mt][0] = __builtin_amdgcn_wmma_f32_16x16x32_f16(false, a, false, bf0, (short)0, acc[mt][0], false, false);
      acc[mt][1] = __builtin_amdgcn_wmma_f32_16x16x32_f16(false, a, false, bf1, (short)0, acc[mt][1], false, false);
    }
  }

  // bias + BN + relu + mean over 49 positions (deterministic shuffle reduction)
  int ch[2] = { nt0 * 16 + mloc, nt1 * 16 + mloc };
  float sch[2], mch[2], bech[2], b1ch[2];
#pragma unroll
  for (int j = 0; j < 2; ++j) {
    sch[j]  = g[ch[j]] * rsqrtf(var[ch[j]] + 1e-5f);
    mch[j]  = mean[ch[j]];
    bech[j] = beta[ch[j]];
    b1ch[j] = b1[ch[j]];
  }
  int m0 = (lane < 16) ? 0 : 8;
  float psum[2] = {0.f, 0.f};
#pragma unroll
  for (int mt = 0; mt < 4; ++mt) {
#pragma unroll
    for (int r = 0; r < 8; ++r) {
      int m = mt * 16 + m0 + r;
      if (m < 49) {
#pragma unroll
        for (int j = 0; j < 2; ++j) {
          float v = acc[mt][j][r] + b1ch[j];
          v = (v - mch[j]) * sch[j] + bech[j];
          v = v > 0.f ? v : 0.f;
          psum[j] += v;
        }
      }
    }
  }
#pragma unroll
  for (int j = 0; j < 2; ++j) {
    float o = __shfl_down(psum[j], 16, 32);
    if (lane < 16) pooled[ch[j]] = psum[j] + o;
  }
  __syncthreads();

  if (tid < oc) {
    float s = b2[tid];
    const float inv = 1.f / 49.f;
    for (int c = 0; c < HC; ++c) s += pooled[c] * inv * w2[tid * HC + c];
    out[n * oc + tid] = s;
  }
}

// ---------------------------------------------------------------------------
__global__ void k_post(const float* __restrict__ s3h, const float* __restrict__ deph,
                       const float* __restrict__ o3h, const float* __restrict__ hdh,
                       const int* __restrict__ classes, const float* __restrict__ rois,
                       const float* __restrict__ mean_size, float* __restrict__ out) {
  int n = blockIdx.x * 256 + threadIdx.x;
  if (n >= NROI) return;
  float* heading = out + OUT_HD;
  float* depth   = out + OUT_DEP;
  float* off3    = out + OUT_O3;
  float* s3o     = out + OUT_S3;
  float* h3o     = out + OUT_H3;
  for (int j = 0; j < 24; ++j) heading[n * 24 + j] = hdh[n * 24 + j];
  float s30 = s3h[n * 4 + 0], s31 = s3h[n * 4 + 1], s32 = s3h[n * 4 + 2], hls = s3h[n * 4 + 3];
  s3o[n * 3 + 0] = s30; s3o[n * 3 + 1] = s31; s3o[n * 3 + 2] = s32;
  h3o[n] = hls;
  off3[n * 2 + 0] = o3h[n * 2 + 0];
  off3[n * 2 + 1] = o3h[n * 2 + 1];
  int cls = classes[n];
  float sz3 = mean_size[cls * 3 + 0] + s30;
  float f_u = rois[n * 20 + 18];
  float b2h = rois[n * 20 + 19];
  float depth_geo = sz3 / b2h * f_u;
  float dgls = hls + 2.f * (logf(f_u) - logf(b2h));
  float d1 = deph[n * 2 + 1];
  float mx = fmaxf(d1, dgls);
  float dls = mx + logf(expf(d1 - mx) + expf(dgls - mx));
  float d0 = deph[n * 2 + 0];
  float sig = 1.f / (1.f + expf(-d0));
  depth[n * 2 + 0] = 1.f / (sig + 1e-6f) - 1.f + depth_geo;
  depth[n * 2 + 1] = dls;
}

// ---------------------------------------------------------------------------
extern "C" void kernel_launch(void* const* d_in, const int* in_sizes, int n_in,
                              void* d_out, int out_size, void* d_ws, size_t ws_size,
                              hipStream_t stream) {
  (void)in_sizes; (void)n_in; (void)out_size; (void)ws_size;
  const float* features     = (const float*)d_in[0];
  const float* calib        = (const float*)d_in[1];
  const float* coord_ranges = (const float*)d_in[2];
  const float* mean_size    = (const float*)d_in[3];
  const float* bw1[3] = { (const float*)d_in[4],  (const float*)d_in[8],  (const float*)d_in[12] };
  const float* bb1[3] = { (const float*)d_in[5],  (const float*)d_in[9],  (const float*)d_in[13] };
  const float* bw2[3] = { (const float*)d_in[6],  (const float*)d_in[10], (const float*)d_in[14] };
  const float* bb2[3] = { (const float*)d_in[7],  (const float*)d_in[11], (const float*)d_in[15] };
  // heads: dep(16..23), o3(24..31), s3(32..39), hd(40..47): w1,b1,g,beta,m,v,w2,b2
  const float* hp[4][8];
  for (int h = 0; h < 4; ++h)
    for (int j = 0; j < 8; ++j) hp[h][j] = (const float*)d_in[16 + h * 8 + j];
  const int hoc[4] = {2, 2, 4, 24};
  float* out = (float*)d_out;

  char* ws = (char*)d_ws;
  size_t off = 0;
  auto take = [&](size_t bytes) { char* p = ws + off; off += (bytes + 255) & ~(size_t)255; return p; };
  _Float16* feat16 = (_Float16*)take((size_t)B_ * H_ * W_ * CIN * 2);
  _Float16* wpm[3];
  for (int i = 0; i < 3; ++i) wpm[i] = (_Float16*)take((size_t)16 * MAIN_KSTEPS * 512 * 2);
  _Float16* wpr[4];
  for (int i = 0; i < 4; ++i) wpr[i] = (_Float16*)take((size_t)16 * ROI_KSTEPS * 512 * 2);
  float* nmsheat = (float*)take((size_t)B_ * 3 * HW * 4);
  float* s_all = (float*)take((size_t)B_ * 3 * KDET * 4);
  int*   i_all = (int*)take((size_t)B_ * 3 * KDET * 4);
  int*   inds  = (int*)take((size_t)B_ * KDET * 4);
  int*   classes = (int*)take((size_t)B_ * KDET * 4);
  float* rois  = (float*)take((size_t)NROI * 20 * 4);
  _Float16* roi_in = (_Float16*)take((size_t)NROI * ROI_SLAB * 2);
  float* hout[4];
  for (int h = 0; h < 4; ++h) hout[h] = (float*)take((size_t)NROI * hoc[h] * 4);

  // 1) features -> NHWC f16
  k_feat_to_f16<<<(B_ * CIN * HW + 255) / 256, 256, 0, stream>>>(features, feat16);
  // 2) pack weights into WMMA B-fragment layout
  {
    int tm = 16 * MAIN_KSTEPS * 512;
    for (int i = 0; i < 3; ++i)
      k_pack_w<<<(tm + 255) / 256, 256, 0, stream>>>(bw1[i], CIN, CIN, MAIN_KSTEPS, wpm[i]);
    int tr = 16 * ROI_KSTEPS * 512;
    for (int h = 0; h < 4; ++h)
      k_pack_w<<<(tr + 255) / 256, 256, 0, stream>>>(hp[h][0], 69, ROI_CPAD, ROI_KSTEPS, wpr[h]);
  }
  // 3) three fused stem branches (WMMA)
  {
    dim3 grid((W_ / PXB) * H_ * B_);
    k_branch<<<grid, 256, 0, stream>>>(feat16, wpm[0], bb1[0], bw2[0], bb2[0], out + OUT_HM, 3);
    k_branch<<<grid, 256, 0, stream>>>(feat16, wpm[1], bb1[1], bw2[1], bb2[1], out + OUT_O2, 2);
    k_branch<<<grid, 256, 0, stream>>>(feat16, wpm[2], bb1[2], bw2[2], bb2[2], out + OUT_S2, 2);
  }
  // 4) NMS + stable top-k
  k_nms<<<(B_ * 3 * HW + 255) / 256, 256, 0, stream>>>(out + OUT_HM, nmsheat);
  k_topk1<<<B_ * 3, 256, 0, stream>>>(nmsheat, s_all, i_all);
  k_topk2<<<B_, 256, 0, stream>>>(s_all, i_all, inds, classes);
  // 5) per-ROI scalars + ROI-align/slab build
  k_rois<<<(NROI + 255) / 256, 256, 0, stream>>>(inds, out + OUT_O2, out + OUT_S2,
                                                 calib, coord_ranges, rois);
  k_roifill<<<NROI, 256, 0, stream>>>(features, rois, classes, roi_in);
  // 6) four fused ROI heads (WMMA)
  for (int h = 0; h < 4; ++h)
    k_roihead<<<NROI, 256, 0, stream>>>(roi_in, wpr[h], hp[h][1], hp[h][2], hp[h][3],
                                        hp[h][4], hp[h][5], hp[h][6], hp[h][7],
                                        hout[h], hoc[h]);
  // 7) postprocess (depth / size3d / heading / offset3d)
  k_post<<<(NROI + 255) / 256, 256, 0, stream>>>(hout[2], hout[0], hout[1], hout[3],
                                                 classes, rois, mean_size, out);
}
